// MultiHeadAttention_6674379178492
// MI455X (gfx1250) — matrix operand order repaired and checked by reading, not compiled, not run on hardware
//
#include <hip/hip_runtime.h>

// ---------------------------------------------------------------------------
// MultiHeadAttention for MI455X (gfx1250, wave32, WMMA).
// All matmuls via v_wmma_f32_16x16x32_bf16; softmax fused (flash-attention).
// GEMM register-blocked 32x64 per wave: 6 fragment loads -> 8 WMMAs per K-step.
// __launch_bounds__(256,1) lifts the VGPR cap so fragments stay in registers
// (round-3 build spilled the loop-invariant Q fragments to scratch).
// ---------------------------------------------------------------------------

typedef __attribute__((ext_vector_type(16))) __bf16 v16bf;
typedef __attribute__((ext_vector_type(8)))  __bf16 v8bf;
typedef __attribute__((ext_vector_type(8)))  float  v8f;

#define D_MODEL  768
#define NUM_HEAD 12
#define HEAD_DIM 64
#define SEQ_N    2048
#define BATCH    2
#define M_ROWS   (BATCH * SEQ_N)          // 4096

static __device__ __forceinline__ v16bf cat8(v8bf lo, v8bf hi) {
    return __builtin_shufflevector(lo, hi, 0, 1, 2, 3, 4, 5, 6, 7,
                                           8, 9, 10, 11, 12, 13, 14, 15);
}

static __device__ __forceinline__ v8f wmma_bf16(v16bf a, v16bf b, v8f c) {
    return __builtin_amdgcn_wmma_f32_16x16x32_bf16(
        /*neg_a=*/false, a, /*neg_b=*/false, b,
        /*c_mod=*/(short)0, c, /*reuse_a=*/false, /*reuse_b=*/false);
}

// ---------------------------------------------------------------------------
// fp32 -> bf16 conversion
// ---------------------------------------------------------------------------
__global__ void cvt_f32_bf16(const float* __restrict__ in,
                             __bf16* __restrict__ out, int n) {
    int i = blockIdx.x * blockDim.x + threadIdx.x;
    if (i < n) out[i] = (__bf16)in[i];
}

// ---------------------------------------------------------------------------
// C[4096 x 768] = A[4096 x 768](bf16) * W[768 x 768](bf16)^T + bias(f32)
// 32x64 output tile per wave; 8 waves per 256-thread block.
// MODE 0: bf16 out, head-split   [B, H, SEQ_N, 64]
// MODE 1: bf16 out, head-split-T [B, H, 64, SEQ_N]   (for V)
// MODE 2: f32  out, plain        [4096, 768]         (final projection)
// ---------------------------------------------------------------------------
template <int MODE>
__global__ __launch_bounds__(256, 1)
void gemm_bf16(const __bf16* __restrict__ A,
               const __bf16* __restrict__ W,
               const float* __restrict__ bias,
               void* __restrict__ outp) {
    constexpr int K    = D_MODEL;
    constexpr int Nout = D_MODEL;

    const int lane = threadIdx.x & 31;
    const int wave = threadIdx.x >> 5;
    const int hl   = lane >> 4;   // half-wave
    const int li   = lane & 15;

    constexpr int ntile = Nout / 64;             // 64-wide column tiles (12)
    const int t  = blockIdx.x * 8 + wave;
    const int tm = (t / ntile) << 5;             // 32 rows
    const int tn = (t % ntile) << 6;             // 64 cols

    const __bf16* ap0 = A + (size_t)(tm + li) * K + hl * 16;
    const __bf16* ap1 = ap0 + (size_t)16 * K;
    const __bf16* bp  = W + (size_t)(tn + li) * K + hl * 16;

    v8f acc[2][4] = {{{}, {}, {}, {}}, {{}, {}, {}, {}}};

    for (int kk = 0; kk < K; kk += 32) {
        // Two A fragments (rows tm..tm+15, tm+16..tm+31).
        v16bf af[2];
        af[0] = cat8(*(const v8bf*)(ap0 + kk), *(const v8bf*)(ap0 + kk + 8));
        af[1] = cat8(*(const v8bf*)(ap1 + kk), *(const v8bf*)(ap1 + kk + 8));
        // Four B fragments (cols tn..tn+63) loaded up front (one clause).
        v16bf bf[4];
#pragma unroll
        for (int c = 0; c < 4; ++c)
            bf[c] = *(const v16bf*)(bp + (size_t)(c * 16) * K + kk);
        // 8 WMMAs back-to-back.
#pragma unroll
        for (int c = 0; c < 4; ++c) {
            acc[0][c] = wmma_bf16(af[0], bf[c], acc[0][c]);
            acc[1][c] = wmma_bf16(af[1], bf[c], acc[1][c]);
        }
    }

#pragma unroll
    for (int c = 0; c < 4; ++c) {
        const int col = tn + c * 16 + li;
        const float bv = bias[col];
#pragma unroll
        for (int mi = 0; mi < 2; ++mi) {
#pragma unroll
            for (int r = 0; r < 8; ++r) {
                const int row = tm + mi * 16 + r + 8 * hl;
                const float val = acc[mi][c][r] + bv;
                if (MODE == 2) {
                    ((float*)outp)[(size_t)row * Nout + col] = val;
                } else {
                    const int b  = row >> 11;        // / SEQ_N
                    const int n  = row & (SEQ_N - 1);
                    const int h  = col >> 6;         // / HEAD_DIM
                    const int dd = col & (HEAD_DIM - 1);
                    __bf16* o = (__bf16*)outp;
                    if (MODE == 0) {
                        o[(((size_t)(b * NUM_HEAD + h) * SEQ_N) + n) * HEAD_DIM +
                          dd] = (__bf16)val;
                    } else {  // MODE 1: transposed (V^T) for attn*V B layout
                        o[(((size_t)(b * NUM_HEAD + h) * HEAD_DIM) + dd) * SEQ_N +
                          n] = (__bf16)val;
                    }
                }
            }
        }
    }
}

// ---------------------------------------------------------------------------
// Flash attention: each wave owns a 16-row query tile for one (b,h),
// streams keys/values 32 at a time. Row statistics live replicated across
// each 16-lane half (C-fragment row layout); reductions via width-16 shuffles.
// ---------------------------------------------------------------------------
__global__ __launch_bounds__(256, 1)
void flash_attn(const __bf16* __restrict__ Qh,
                const __bf16* __restrict__ Kh,
                const __bf16* __restrict__ Vt,
                const int* __restrict__ mask,
                __bf16* __restrict__ Obuf) {
    __shared__ __bf16 plds[8][16 * 32];   // per-wave P tile (16 rows x 32 keys)

    const int lane = threadIdx.x & 31;
    const int wave = threadIdx.x >> 5;
    const int hl   = lane >> 4;
    const int li   = lane & 15;

    constexpr int blocks_per_bh = SEQ_N / 128;     // 8 waves x 16 rows
    const int bh   = blockIdx.x / blocks_per_bh;
    const int qblk = blockIdx.x % blocks_per_bh;
    const int b = bh / NUM_HEAD;
    const int h = bh % NUM_HEAD;
    const int q0 = qblk * 128 + wave * 16;

    const __bf16* Qp = Qh + (size_t)bh * SEQ_N * HEAD_DIM;
    const __bf16* Kp = Kh + (size_t)bh * SEQ_N * HEAD_DIM;
    const __bf16* Vp = Vt + (size_t)bh * HEAD_DIM * SEQ_N;   // [64][SEQ_N]
    const int*    mp = mask + (size_t)b * SEQ_N * SEQ_N;

    // Preload the two Q A-fragments (d = 0..31 and 32..63).
    v16bf aq[2];
    {
        const int row = q0 + li;
#pragma unroll
        for (int t = 0; t < 2; ++t) {
            const __bf16* p = Qp + (size_t)row * HEAD_DIM + t * 32 + hl * 16;
            aq[t] = cat8(*(const v8bf*)p, *(const v8bf*)(p + 8));
        }
    }

    v8f o[4] = {{}, {}, {}, {}};          // 16x64 output accumulator
    float mrow[8], lrow[8];
#pragma unroll
    for (int r = 0; r < 8; ++r) { mrow[r] = -3.0e38f; lrow[r] = 0.0f; }

    const float scale = 0.125f;           // 1/sqrt(64)

    for (int j = 0; j < SEQ_N; j += 32) {
        // ---- scores S[16 x 32] = Q * K^T ----
        v8f s0 = {}, s1 = {};
        {
            v16bf kb[4];
#pragma unroll
            for (int t = 0; t < 2; ++t) {
                kb[t]     = *(const v16bf*)(Kp + (size_t)(j + li) * HEAD_DIM + t * 32 + hl * 16);
                kb[2 + t] = *(const v16bf*)(Kp + (size_t)(j + 16 + li) * HEAD_DIM + t * 32 + hl * 16);
            }
#pragma unroll
            for (int t = 0; t < 2; ++t) {
                s0 = wmma_bf16(aq[t], kb[t], s0);
                s1 = wmma_bf16(aq[t], kb[2 + t], s1);
            }
        }

        // ---- scale + mask ----
#pragma unroll
        for (int r = 0; r < 8; ++r) {
            const int row = q0 + r + 8 * hl;
            const int m0 = mp[(size_t)row * SEQ_N + j + li];
            const int m1 = mp[(size_t)row * SEQ_N + j + 16 + li];
            const float x0 = s0[r] * scale;
            const float x1 = s1[r] * scale;
            s0[r] = m0 ? x0 : -1.0e9f;
            s1[r] = m1 ? x1 : -1.0e9f;
        }

        // ---- online softmax (per-row stats replicated over 16-lane halves) ----
        float alpha[8];
#pragma unroll
        for (int r = 0; r < 8; ++r) {
            float mx = fmaxf(s0[r], s1[r]);
#pragma unroll
            for (int d = 1; d < 16; d <<= 1) mx = fmaxf(mx, __shfl_xor(mx, d, 16));
            const float mnew = fmaxf(mrow[r], mx);
            const float a = __expf(mrow[r] - mnew);
            mrow[r] = mnew;
            alpha[r] = a;
            const float p0 = __expf(s0[r] - mnew);
            const float p1 = __expf(s1[r] - mnew);
            s0[r] = p0;
            s1[r] = p1;
            float sum = p0 + p1;
#pragma unroll
            for (int d = 1; d < 16; d <<= 1) sum += __shfl_xor(sum, d, 16);
            lrow[r] = lrow[r] * a + sum;
        }
#pragma unroll
        for (int t = 0; t < 4; ++t)
#pragma unroll
            for (int r = 0; r < 8; ++r) o[t][r] *= alpha[r];

        // ---- P: C-fragment layout -> A-fragment layout via per-wave LDS ----
        __bf16* pl = plds[wave];
#pragma unroll
        for (int r = 0; r < 8; ++r) {
            const int rl = r + 8 * hl;
            pl[rl * 32 + li]      = (__bf16)s0[r];
            pl[rl * 32 + 16 + li] = (__bf16)s1[r];
        }
        __syncthreads();
        v16bf pa;
        {
            const __bf16* pr = pl + li * 32 + hl * 16;
            pa = cat8(*(const v8bf*)pr, *(const v8bf*)(pr + 8));
        }
        __syncthreads();

        // ---- O += P * V  (V already stored [dd][m], native B layout) ----
        v16bf vb[4];
#pragma unroll
        for (int t = 0; t < 4; ++t)
            vb[t] = *(const v16bf*)(Vp + (size_t)(t * 16 + li) * SEQ_N + j + hl * 16);
#pragma unroll
        for (int t = 0; t < 4; ++t)
            o[t] = wmma_bf16(pa, vb[t], o[t]);
    }

    // ---- finalize: O /= l, write merged-head bf16 [B, SEQ_N, D_MODEL] ----
#pragma unroll
    for (int r = 0; r < 8; ++r) {
        const int row = q0 + r + 8 * hl;
        const float inv = 1.0f / lrow[r];
        const size_t base =
            ((size_t)(b * SEQ_N) + row) * D_MODEL + h * HEAD_DIM + li;
#pragma unroll
        for (int t = 0; t < 4; ++t)
            Obuf[base + t * 16] = (__bf16)(o[t][r] * inv);
    }
}

// ---------------------------------------------------------------------------
// Launch
// ---------------------------------------------------------------------------
extern "C" void kernel_launch(void* const* d_in, const int* in_sizes, int n_in,
                              void* d_out, int out_size, void* d_ws, size_t ws_size,
                              hipStream_t stream) {
    (void)in_sizes; (void)n_in; (void)out_size; (void)ws_size;

    const float* q    = (const float*)d_in[0];
    const float* k    = (const float*)d_in[1];
    const float* v    = (const float*)d_in[2];
    const int*   mask = (const int*)  d_in[3];
    const float* wq   = (const float*)d_in[4];
    const float* bq   = (const float*)d_in[5];
    const float* wk   = (const float*)d_in[6];
    const float* bk   = (const float*)d_in[7];
    const float* wv   = (const float*)d_in[8];
    const float* bv   = (const float*)d_in[9];
    const float* wo   = (const float*)d_in[10];
    const float* bo   = (const float*)d_in[11];

    const size_t actElems = (size_t)M_ROWS * D_MODEL;      // 3,145,728
    const size_t wElems   = (size_t)D_MODEL * D_MODEL;     // 589,824

    char* ws = (char*)d_ws;
    __bf16* Xq  = (__bf16*)ws; ws += actElems * 2;
    __bf16* Xk  = (__bf16*)ws; ws += actElems * 2;
    __bf16* Xv  = (__bf16*)ws; ws += actElems * 2;
    __bf16* Wq  = (__bf16*)ws; ws += wElems * 2;
    __bf16* Wk  = (__bf16*)ws; ws += wElems * 2;
    __bf16* Wv  = (__bf16*)ws; ws += wElems * 2;
    __bf16* Wo  = (__bf16*)ws; ws += wElems * 2;
    __bf16* Qh  = (__bf16*)ws; ws += actElems * 2;   // [B,H,N,64]
    __bf16* Kh  = (__bf16*)ws; ws += actElems * 2;   // [B,H,N,64]
    __bf16* Vt  = (__bf16*)ws; ws += actElems * 2;   // [B,H,64,N]
    __bf16* Ob  = (__bf16*)ws; ws += actElems * 2;   // [B*N, D_MODEL]

    // fp32 -> bf16
    {
        const int na = (int)actElems, nw = (int)wElems;
        const int ba = (na + 255) / 256, bw = (nw + 255) / 256;
        cvt_f32_bf16<<<ba, 256, 0, stream>>>(q,  Xq, na);
        cvt_f32_bf16<<<ba, 256, 0, stream>>>(k,  Xk, na);
        cvt_f32_bf16<<<ba, 256, 0, stream>>>(v,  Xv, na);
        cvt_f32_bf16<<<bw, 256, 0, stream>>>(wq, Wq, nw);
        cvt_f32_bf16<<<bw, 256, 0, stream>>>(wk, Wk, nw);
        cvt_f32_bf16<<<bw, 256, 0, stream>>>(wv, Wv, nw);
        cvt_f32_bf16<<<bw, 256, 0, stream>>>(wo, Wo, nw);
    }

    // Q/K/V projections (WMMA), head-split outputs (V transposed)
    // 32x64 tile per wave -> (M/32)*(D_MODEL/64) wave-tiles, 8 waves/block.
    const int tiles = (M_ROWS / 32) * (D_MODEL / 64);   // 1536
    const int gblk  = tiles / 8;                        // 192
    gemm_bf16<0><<<gblk, 256, 0, stream>>>(Xq, Wq, bq, Qh);
    gemm_bf16<0><<<gblk, 256, 0, stream>>>(Xk, Wk, bk, Kh);
    gemm_bf16<1><<<gblk, 256, 0, stream>>>(Xv, Wv, bv, Vt);

    // fused masked-softmax attention
    const int ablk = BATCH * NUM_HEAD * (SEQ_N / 128);  // 384
    flash_attn<<<ablk, 256, 0, stream>>>(Qh, Kh, Vt, mask, Ob);

    // output projection, fp32 result
    gemm_bf16<2><<<gblk, 256, 0, stream>>>(Ob, Wo, bo, d_out);
}
